// Decoder_5695126634618
// MI455X (gfx1250) — matrix operand order repaired and checked by reading, not compiled, not run on hardware
//
#include <hip/hip_runtime.h>
#include <hip/hip_bf16.h>

// ---------------------------------------------------------------------------
// Decoder step: attention (Bahdanau) + 2-layer LSTM + output projection.
// B=64, S=1024, V=32000, D=H=E=A=1024, L=2.
//
// MI455X plan:
//  * encoder_outputs fp32 (268MB) -> bf16 ws copy (134MB, L2-resident) once;
//    the dominant batched GEMM re-reads panels from L2 instead of HBM.
//  * k-GEMM (137 GFLOP) on v_wmma_f32_16x16x32_bf16, double-buffered LDS
//    staging via GLOBAL_LOAD_ASYNC_TO_LDS_B128 (+ s_wait_asynccnt), fused
//    epilogue sum_a tanh(q+k)*va (k never materialized).
//  * LDS rows strided 40 shorts (80B): per-lane WMMA fragments are 16B-aligned
//    contiguous runs -> ds_load_b128 x2 per fragment, conflict-free banks.
//  * q / LSTM gates / logits share one M=64 WMMA GEMM (fp32 -> bf16 on the
//    fly; weight panels are read exactly once).
//  * bva dropped (softmax-invariant); ba_e folded into q's bias.
// ---------------------------------------------------------------------------

#define BB 64
#define SS 1024
#define VV 32000
#define DD 1024
#define HH 1024
#define EE 1024
#define AA 1024
#define LDA 40   // LDS row stride in shorts: 80B = 5*16B, 16B-aligned rows

typedef __attribute__((ext_vector_type(16))) __bf16 v16bf;
typedef __attribute__((ext_vector_type(8)))  float  v8f;
typedef __attribute__((ext_vector_type(8)))  unsigned short us8;  // 16B

union FragU { us8 h[2]; v16bf v; };

__device__ __forceinline__ unsigned short f2bf(float f) {
  unsigned u = __float_as_uint(f);
  u += 0x7FFFu + ((u >> 16) & 1u);            // round-to-nearest-even
  return (unsigned short)(u >> 16);
}
__device__ __forceinline__ float bf2f(unsigned short h) {
  return __uint_as_float(((unsigned)h) << 16);
}
__device__ __forceinline__ float sigmoidf(float x) {
  return 1.0f / (1.0f + __expf(-x));
}

// Async global->LDS copy, 16B per lane. VDST = LDS byte address (low 32 bits
// of the generic pointer, ISA 10.2), VADDR = 64-bit global address, tracked
// by ASYNCcnt.
__device__ __forceinline__ void async_copy_b128(void* lds, const void* g) {
  const unsigned lo = (unsigned)(unsigned long long)lds;
  const unsigned long long ga = (unsigned long long)g;
  asm volatile("global_load_async_to_lds_b128 %0, %1, off"
               :: "v"(lo), "v"(ga) : "memory");
}
__device__ __forceinline__ void wait_async0() {
  asm volatile("s_wait_asynccnt 0" ::: "memory");
}

// A-operand fragment (16x32 bf16). Per ISA: lane l -> M=l&15, g=l>>4;
// u[0..7] = As[m][8g .. 8g+7], u[8..15] = As[m][16+8g .. 16+8g+7]
// -> two aligned 16B LDS vector loads.
__device__ __forceinline__ v16bf load_a_frag(const unsigned short* base, int lane) {
  const int g = lane >> 4, m = lane & 15;
  const unsigned short* r = base + m * LDA + 8 * g;
  FragU f;
  f.h[0] = *(const us8*)(r);
  f.h[1] = *(const us8*)(r + 16);
  return f.v;
}
// B-operand fragment (32x16 bf16, Bs holds W rows [n][k]): lane l -> N=l&15;
// u[0..7] = Bs[n][8g .. 8g+7], u[8..15] = Bs[n][16+8g .. 16+8g+7].
__device__ __forceinline__ v16bf load_b_frag(const unsigned short* base, int lane) {
  const int g = lane >> 4, n = lane & 15;
  const unsigned short* r = base + n * LDA + 8 * g;
  FragU f;
  f.h[0] = *(const us8*)(r);
  f.h[1] = *(const us8*)(r + 16);
  return f.v;
}

// ---------------------------------------------------------------------------
// fp32 -> bf16 streaming conversion (4 elems/thread/iter)
// ---------------------------------------------------------------------------
__global__ __launch_bounds__(256) void cvt_bf16_kernel(const float* __restrict__ src,
                                                       unsigned short* __restrict__ dst,
                                                       long n4) {
  long i = (long)blockIdx.x * blockDim.x + threadIdx.x;
  const long stride = (long)gridDim.x * blockDim.x;
  for (; i < n4; i += stride) {
    float4 f = ((const float4*)src)[i];
    uint2 o;
    o.x = (unsigned)f2bf(f.x) | ((unsigned)f2bf(f.y) << 16);
    o.y = (unsigned)f2bf(f.z) | ((unsigned)f2bf(f.w) << 16);
    ((uint2*)dst)[i] = o;
  }
}

// ---------------------------------------------------------------------------
// prep: zero score[B*S], gather embedding rows into x[:, 0:D]
// ---------------------------------------------------------------------------
__global__ __launch_bounds__(256) void prep_kernel(const int* __restrict__ input_step,
                                                   const float* __restrict__ emb,
                                                   float* __restrict__ score,
                                                   float* __restrict__ x) {
  const int i = blockIdx.x * 256 + threadIdx.x;     // 0..65535
  score[i] = 0.0f;
  const int b = i >> 10, d = i & 1023;
  x[b * (DD + EE) + d] = emb[(long)input_step[b] * DD + d];
}

// ---------------------------------------------------------------------------
// Generic M=64 GEMM:  C(64xN) = A0(64xK0) @ W0^T + A1(64xK1) @ W1^T + b0 + b1
// W row-major (N x K). fp32 in/out, bf16 WMMA. One 64x128 tile/block, 8 waves.
// ---------------------------------------------------------------------------
__global__ __launch_bounds__(256)
void gemm64_kernel(const float* __restrict__ A0, const float* __restrict__ W0, int K0,
                   const float* __restrict__ A1, const float* __restrict__ W1, int K1,
                   const float* __restrict__ b0, const float* __restrict__ b1,
                   float* __restrict__ C, int N) {
  __shared__ __align__(16) unsigned short As[64 * LDA];
  __shared__ __align__(16) unsigned short Bs[128 * LDA];
  const int t = threadIdx.x;
  const int lane = t & 31, wave = t >> 5;
  const int wm = wave & 3;        // 16-row M subtile
  const int wn = wave >> 2;       // 64-col N half
  const v8f vzero = {0.f, 0.f, 0.f, 0.f, 0.f, 0.f, 0.f, 0.f};
  v8f acc[4] = {vzero, vzero, vzero, vzero};

  for (int phase = 0; phase < 2; ++phase) {
    const float* A = phase ? A1 : A0;
    const float* W = phase ? W1 : W0;
    const int K = phase ? K1 : K0;
    if (A == nullptr || K == 0) continue;
    for (int kk = 0; kk < K; kk += 32) {
      {   // stage A tile 64x32: 8 floats/thread -> one 16B LDS store
        const int row = t >> 2, ko = (t & 3) * 8;
        const float* src = A + (long)row * K + kk + ko;
        float4 f0 = ((const float4*)src)[0];
        float4 f1 = ((const float4*)src)[1];
        us8 v;
        v[0] = f2bf(f0.x); v[1] = f2bf(f0.y); v[2] = f2bf(f0.z); v[3] = f2bf(f0.w);
        v[4] = f2bf(f1.x); v[5] = f2bf(f1.y); v[6] = f2bf(f1.z); v[7] = f2bf(f1.w);
        *(us8*)&As[row * LDA + ko] = v;
      }
      {   // stage W tile 128x32: 16 floats/thread -> two 16B LDS stores
        const int row = t >> 1, ko = (t & 1) * 16;
        const float* src = W + (long)(blockIdx.x * 128 + row) * K + kk + ko;
        if (kk + 32 < K) __builtin_prefetch(src + 32, 0, 0);   // next K panel
        float4 f0 = ((const float4*)src)[0];
        float4 f1 = ((const float4*)src)[1];
        float4 f2 = ((const float4*)src)[2];
        float4 f3 = ((const float4*)src)[3];
        us8 vlo, vhi;
        vlo[0] = f2bf(f0.x); vlo[1] = f2bf(f0.y); vlo[2] = f2bf(f0.z); vlo[3] = f2bf(f0.w);
        vlo[4] = f2bf(f1.x); vlo[5] = f2bf(f1.y); vlo[6] = f2bf(f1.z); vlo[7] = f2bf(f1.w);
        vhi[0] = f2bf(f2.x); vhi[1] = f2bf(f2.y); vhi[2] = f2bf(f2.z); vhi[3] = f2bf(f2.w);
        vhi[4] = f2bf(f3.x); vhi[5] = f2bf(f3.y); vhi[6] = f2bf(f3.z); vhi[7] = f2bf(f3.w);
        *(us8*)&Bs[row * LDA + ko] = vlo;
        *(us8*)&Bs[row * LDA + ko + 8] = vhi;
      }
      __syncthreads();
      const v16bf af = load_a_frag(&As[wm * 16 * LDA], lane);
#pragma unroll
      for (int ni = 0; ni < 4; ++ni) {
        const v16bf bfg = load_b_frag(&Bs[(wn * 64 + ni * 16) * LDA], lane);
        acc[ni] = __builtin_amdgcn_wmma_f32_16x16x32_bf16(
            false, af, false, bfg, (short)0, acc[ni], false, false);
      }
      __syncthreads();
    }
  }
  // C layout: lane l holds N = l&15; accumulator elem r holds M = r + 8*(l>>4)
  const int g = lane >> 4, nlo = lane & 15;
#pragma unroll
  for (int ni = 0; ni < 4; ++ni) {
    const int n = blockIdx.x * 128 + wn * 64 + ni * 16 + nlo;
    const float bias = (b0 ? b0[n] : 0.0f) + (b1 ? b1[n] : 0.0f);
#pragma unroll
    for (int r = 0; r < 8; ++r) {
      const int m = wm * 16 + g * 8 + r;
      C[(long)m * N + n] = acc[ni][r] + bias;
    }
  }
}

// ---------------------------------------------------------------------------
// Batched attention-energy GEMM with fused tanh/dot-va epilogue.
// grid(A/128, S/128, B); per block: 128x128 k-tile, K=E=1024, double-buffered
// LDS staged with GLOBAL_LOAD_ASYNC_TO_LDS_B128 when sources are bf16.
// score[b][s] += sum_a tanh(q'[b][a] + k[b][s][a]) * va[a]
// ---------------------------------------------------------------------------
template <bool BF>
__global__ __launch_bounds__(256)
void attn_score_kernel(const void* __restrict__ encp, const void* __restrict__ waep,
                       const float* __restrict__ q, const float* __restrict__ va,
                       float* __restrict__ score) {
  __shared__ __align__(16) unsigned short As[2][128 * LDA];
  __shared__ __align__(16) unsigned short Bs[2][128 * LDA];
  __shared__ float sscore[128];
  const int t = threadIdx.x, lane = t & 31, wave = t >> 5;
  const int wm = wave & 3;      // 32-row M (s) slab -> 2 subtiles
  const int wn = wave >> 2;     // 64-col N (a) half -> 4 subtiles
  const int b = blockIdx.z;
  const long encBase = ((long)b * SS + (long)blockIdx.y * 128) * EE;
  const long waeBase = (long)blockIdx.x * 128 * EE;
  const v8f vzero = {0.f, 0.f, 0.f, 0.f, 0.f, 0.f, 0.f, 0.f};
  v8f acc[2][4] = {{vzero, vzero, vzero, vzero}, {vzero, vzero, vzero, vzero}};
  if (t < 128) sscore[t] = 0.0f;

  const int row = t >> 1, ko = (t & 1) * 16;       // 16 shorts (32B) / thread / tile
  auto stage = [&](int buf, int kk) {
    unsigned short* da = &As[buf][row * LDA + ko];
    unsigned short* db = &Bs[buf][row * LDA + ko];
    if (BF) {
      const unsigned short* sa = (const unsigned short*)encp + encBase + (long)row * EE + kk + ko;
      const unsigned short* sb = (const unsigned short*)waep + waeBase + (long)row * EE + kk + ko;
      async_copy_b128(da,     sa);
      async_copy_b128(da + 8, sa + 8);
      async_copy_b128(db,     sb);
      async_copy_b128(db + 8, sb + 8);
    } else {
      const float* sa = (const float*)encp + encBase + (long)row * EE + kk + ko;
      const float* sb = (const float*)waep + waeBase + (long)row * EE + kk + ko;
      us8 a0, a1, b0, b1;
#pragma unroll
      for (int i = 0; i < 8; ++i) {
        a0[i] = f2bf(sa[i]); a1[i] = f2bf(sa[i + 8]);
        b0[i] = f2bf(sb[i]); b1[i] = f2bf(sb[i + 8]);
      }
      *(us8*)da = a0; *(us8*)(da + 8) = a1;
      *(us8*)db = b0; *(us8*)(db + 8) = b1;
    }
  };

  stage(0, 0);
  if (BF) wait_async0();
  __syncthreads();
  for (int kk = 0; kk < EE; kk += 32) {
    const int cur = (kk >> 5) & 1;
    if (kk + 32 < EE) stage(cur ^ 1, kk + 32);   // prefetch next tile (async)
#pragma unroll
    for (int mi = 0; mi < 2; ++mi) {
      const v16bf af = load_a_frag(&As[cur][(wm * 32 + mi * 16) * LDA], lane);
#pragma unroll
      for (int ni = 0; ni < 4; ++ni) {
        const v16bf bfg = load_b_frag(&Bs[cur][(wn * 64 + ni * 16) * LDA], lane);
        acc[mi][ni] = __builtin_amdgcn_wmma_f32_16x16x32_bf16(
            false, af, false, bfg, (short)0, acc[mi][ni], false, false);
      }
    }
    if (BF) wait_async0();    // next buffer's async copies complete
    __syncthreads();          // all waves done reading cur / writing next
  }

  // Fused epilogue: v = tanh(q + k)*va, 16-lane shuffle reduce (one s row each)
  const int g = lane >> 4, nlo = lane & 15;
#pragma unroll
  for (int ni = 0; ni < 4; ++ni) {
    const int a = blockIdx.x * 128 + wn * 64 + ni * 16 + nlo;
    const float qa = q[b * AA + a];
    const float vaa = va[a];
#pragma unroll
    for (int mi = 0; mi < 2; ++mi) {
#pragma unroll
      for (int r = 0; r < 8; ++r) {
        float v = tanhf(qa + acc[mi][ni][r]) * vaa;
        v += __shfl_xor(v, 1); v += __shfl_xor(v, 2);
        v += __shfl_xor(v, 4); v += __shfl_xor(v, 8);
        if (nlo == 0) atomicAdd(&sscore[wm * 32 + mi * 16 + g * 8 + r], v);
      }
    }
  }
  __syncthreads();
  if (t < 128) atomicAdd(&score[(long)b * SS + blockIdx.y * 128 + t], sscore[t]);
}

// ---------------------------------------------------------------------------
// softmax over S per batch row (bva is softmax-invariant -> omitted)
// ---------------------------------------------------------------------------
__global__ __launch_bounds__(256) void softmax_kernel(const float* __restrict__ score,
                                                      float* __restrict__ attn) {
  __shared__ float red[256];
  const int b = blockIdx.x, t = threadIdx.x;
  float loc[4], m = -3.402823466e38f;
#pragma unroll
  for (int j = 0; j < 4; ++j) { loc[j] = score[b * SS + t + j * 256]; m = fmaxf(m, loc[j]); }
  red[t] = m; __syncthreads();
  for (int s = 128; s > 0; s >>= 1) { if (t < s) red[t] = fmaxf(red[t], red[t + s]); __syncthreads(); }
  m = red[0]; __syncthreads();
  float sum = 0.0f;
#pragma unroll
  for (int j = 0; j < 4; ++j) { loc[j] = __expf(loc[j] - m); sum += loc[j]; }
  red[t] = sum; __syncthreads();
  for (int s = 128; s > 0; s >>= 1) { if (t < s) red[t] += red[t + s]; __syncthreads(); }
  const float inv = 1.0f / red[0];
#pragma unroll
  for (int j = 0; j < 4; ++j) attn[b * SS + t + j * 256] = loc[j] * inv;
}

// ---------------------------------------------------------------------------
// context[b][e] = sum_s attn[b][s] * enc[b][s][e]  -> x[:, D:D+E]
// ---------------------------------------------------------------------------
template <bool BF>
__global__ __launch_bounds__(256) void context_kernel(const void* __restrict__ encp,
                                                      const float* __restrict__ attn,
                                                      float* __restrict__ x) {
  const int b = blockIdx.x, t = threadIdx.x;
  const float* ab = attn + b * SS;
  float acc[4] = {0.f, 0.f, 0.f, 0.f};
  if (BF) {
    const unsigned short* eb = (const unsigned short*)encp + (long)b * SS * EE;
    for (int s = 0; s < SS; ++s) {
      const float w = ab[s];
#pragma unroll
      for (int j = 0; j < 4; ++j) acc[j] += w * bf2f(eb[s * EE + t + j * 256]);
    }
  } else {
    const float* eb = (const float*)encp + (long)b * SS * EE;
    for (int s = 0; s < SS; ++s) {
      const float w = ab[s];
#pragma unroll
      for (int j = 0; j < 4; ++j) acc[j] += w * eb[s * EE + t + j * 256];
    }
  }
#pragma unroll
  for (int j = 0; j < 4; ++j) x[b * (DD + EE) + DD + t + j * 256] = acc[j];
}

// ---------------------------------------------------------------------------
// LSTM cell pointwise: gates (B x 4H, order i|f|g|o) + c_prev -> h_out, c_out
// ---------------------------------------------------------------------------
__global__ __launch_bounds__(256) void lstm_elem_kernel(const float* __restrict__ gates,
                                                        const float* __restrict__ cprev,
                                                        float* __restrict__ hout,
                                                        float* __restrict__ cout) {
  const int i = blockIdx.x * 256 + threadIdx.x;     // 0..B*H-1
  const int b = i >> 10, hh = i & 1023;
  const float* gb = gates + b * 4 * HH;
  const float ig = sigmoidf(gb[hh]);
  const float fg = sigmoidf(gb[HH + hh]);
  const float gg = tanhf(gb[2 * HH + hh]);
  const float og = sigmoidf(gb[3 * HH + hh]);
  const float c = fg * cprev[i] + ig * gg;
  cout[i] = c;
  hout[i] = og * tanhf(c);
}

// ---------------------------------------------------------------------------
// launcher
// ---------------------------------------------------------------------------
extern "C" void kernel_launch(void* const* d_in, const int* in_sizes, int n_in,
                              void* d_out, int out_size, void* d_ws, size_t ws_size,
                              hipStream_t stream) {
  (void)in_sizes; (void)n_in; (void)out_size;
  const int*   input_step = (const int*)  d_in[0];
  const float* h0   = (const float*)d_in[1];   // (2,B,H)
  const float* c0   = (const float*)d_in[2];   // (2,B,H)
  const float* enc  = (const float*)d_in[3];   // (B,S,E)
  const float* emb  = (const float*)d_in[4];   // (V,D)
  const float* Wa_h = (const float*)d_in[5];
  const float* ba_h = (const float*)d_in[6];
  const float* Wa_e = (const float*)d_in[7];
  const float* ba_e = (const float*)d_in[8];
  const float* va   = (const float*)d_in[9];
  // d_in[10] = bva: scalar shift, softmax-invariant -> unused
  const float* Wih0 = (const float*)d_in[11];
  const float* Whh0 = (const float*)d_in[12];
  const float* bih0 = (const float*)d_in[13];
  const float* bhh0 = (const float*)d_in[14];
  const float* Wih1 = (const float*)d_in[15];
  const float* Whh1 = (const float*)d_in[16];
  const float* bih1 = (const float*)d_in[17];
  const float* bhh1 = (const float*)d_in[18];
  const float* Wout = (const float*)d_in[19];
  const float* bout = (const float*)d_in[20];
  float* out = (float*)d_out;

  // d_out layout (floats): logits | h_new(2,B,H) | c_new(2,B,H) | attn(B,S)
  const long LOGITS_N = (long)BB * VV;                  // 2,048,000
  const long H_OFF = LOGITS_N;
  const long C_OFF = H_OFF + 2L * BB * HH;
  const long ATTN_OFF = C_OFF + 2L * BB * HH;

  // workspace carve-out
  const size_t ENC_BF = (size_t)BB * SS * EE * 2;       // 134,217,728 B
  const size_t WAE_BF = (size_t)AA * EE * 2;            // 2,097,152 B
  const size_t SMALL  = (size_t)(BB * AA + BB * SS + BB * (DD + EE) + BB * 4 * HH) * 4;
  const bool use_bf = (ws_size >= ENC_BF + WAE_BF + SMALL);

  char* p = (char*)d_ws;
  unsigned short* enc_bf = nullptr;
  unsigned short* wae_bf = nullptr;
  if (use_bf) { enc_bf = (unsigned short*)p; p += ENC_BF;
                wae_bf = (unsigned short*)p; p += WAE_BF; }
  float* q     = (float*)p; p += (size_t)BB * AA * 4;
  float* score = (float*)p; p += (size_t)BB * SS * 4;
  float* x     = (float*)p; p += (size_t)BB * (DD + EE) * 4;
  float* gates = (float*)p;

  // 1) L2-resident bf16 operand copies
  if (use_bf) {
    cvt_bf16_kernel<<<4096, 256, 0, stream>>>(enc, enc_bf, (long)BB * SS * EE / 4);
    cvt_bf16_kernel<<<256, 256, 0, stream>>>(Wa_e, wae_bf, (long)AA * EE / 4);
  }
  // 2) zero score + embedding gather into x[:, :D]
  prep_kernel<<<256, 256, 0, stream>>>(input_step, emb, score, x);
  // 3) q' = h0[-1] @ Wa_h^T + ba_h + ba_e   (ba_e folded here)
  gemm64_kernel<<<AA / 128, 256, 0, stream>>>(h0 + (long)BB * HH, Wa_h, HH,
                                              nullptr, nullptr, 0,
                                              ba_h, ba_e, q, AA);
  // 4) fused energy GEMM + tanh·va reduction -> score
  if (use_bf)
    attn_score_kernel<true ><<<dim3(AA / 128, SS / 128, BB), 256, 0, stream>>>(enc_bf, wae_bf, q, va, score);
  else
    attn_score_kernel<false><<<dim3(AA / 128, SS / 128, BB), 256, 0, stream>>>(enc, Wa_e, q, va, score);
  // 5) softmax -> attn output
  softmax_kernel<<<BB, 256, 0, stream>>>(score, out + ATTN_OFF);
  // 6) context -> x[:, D:]
  if (use_bf) context_kernel<true ><<<BB, 256, 0, stream>>>(enc_bf, out + ATTN_OFF, x);
  else        context_kernel<false><<<BB, 256, 0, stream>>>(enc,    out + ATTN_OFF, x);
  // 7) layer-0 gates = x@Wih0^T + h0[0]@Whh0^T + biases ; cell -> h1,c1
  gemm64_kernel<<<4 * HH / 128, 256, 0, stream>>>(x, Wih0, DD + EE, h0, Whh0, HH,
                                                  bih0, bhh0, gates, 4 * HH);
  lstm_elem_kernel<<<BB * HH / 256, 256, 0, stream>>>(gates, c0, out + H_OFF, out + C_OFF);
  // 8) layer-1 gates = h1@Wih1^T + h0[1]@Whh1^T ; cell -> h2,c2
  gemm64_kernel<<<4 * HH / 128, 256, 0, stream>>>(out + H_OFF, Wih1, HH,
                                                  h0 + (long)BB * HH, Whh1, HH,
                                                  bih1, bhh1, gates, 4 * HH);
  lstm_elem_kernel<<<BB * HH / 256, 256, 0, stream>>>(gates, c0 + (long)BB * HH,
                                                      out + H_OFF + (long)BB * HH,
                                                      out + C_OFF + (long)BB * HH);
  // 9) logits = h2 @ Wout^T + bout  (straight into d_out)
  gemm64_kernel<<<VV / 128, 256, 0, stream>>>(out + H_OFF + (long)BB * HH, Wout, HH,
                                              nullptr, nullptr, 0,
                                              bout, nullptr, out, VV);
}
